// TEST_MODEL_33208687133417
// MI455X (gfx1250) — matrix operand order repaired and checked by reading, not compiled, not run on hardware
//
#include <hip/hip_runtime.h>

typedef _Float16 half_t;
typedef __attribute__((ext_vector_type(8)))  _Float16 v8h;
typedef __attribute__((ext_vector_type(16))) _Float16 v16h;
typedef __attribute__((ext_vector_type(8)))  float    v8f;

#define N_NODES    50000
#define N_EDGES    500000
#define D          128
#define K_CLUSTERS 1024
#define EPS        1e-8f

// ---------- WMMA tile loaders (contiguous-k layouts, k = 16*(lane>>4)+s, wave32) ----------
// A-matrix 16x32 f16, absolute row form: lane holds the given row; halves
// 0..15 = K (lane>>4)*16+{0..15} -> two 16-byte loads.
__device__ __forceinline__ v16h load_tileA_row(const half_t* base, int ld,
                                               int row, int k0, int lane) {
    int koff = k0 + ((lane >> 4) << 4);
    const half_t* p = base + (size_t)row * ld + koff;
    v8h lo = *(const v8h*)(p);
    v8h hi = *(const v8h*)(p + 8);
    return __builtin_shufflevector(lo, hi, 0,1,2,3,4,5,6,7,8,9,10,11,12,13,14,15);
}

// B-matrix 32x16 f16 (B[k][n] = rowmajor_src[n][k]): lane L holds col (L&15);
// halves 0..15 = K (L>>4)*16 + {0..15} -> one contiguous 32-byte load.
__device__ __forceinline__ v16h load_tileB32x16(const half_t* base, int ld,
                                                int col0, int k0, int lane) {
    int n = col0 + (lane & 15);
    int k = k0 + ((lane >> 4) << 4);
    const v8h* p = (const v8h*)(base + (size_t)n * ld + k);
    v8h lo = p[0];
    v8h hi = p[1];
    return __builtin_shufflevector(lo, hi, 0,1,2,3,4,5,6,7,8,9,10,11,12,13,14,15);
}

// ---------- 1) edge scatter-add: agg[dst] += x[src], deg[dst] += 1 ----------
__global__ void __launch_bounds__(256)
agg_kernel(const float* __restrict__ x, const int* __restrict__ ei,
           float* __restrict__ agg, float* __restrict__ deg) {
    int wave = (blockIdx.x * blockDim.x + threadIdx.x) >> 5;
    int lane = threadIdx.x & 31;
    if (wave >= N_EDGES) return;
    int src = ei[wave];
    int dst = ei[N_EDGES + wave];
    const float4 v = *(const float4*)(x + (size_t)src * D + lane * 4);
    float* a = agg + (size_t)dst * D + lane * 4;
    unsafeAtomicAdd(a + 0, v.x);
    unsafeAtomicAdd(a + 1, v.y);
    unsafeAtomicAdd(a + 2, v.z);
    unsafeAtomicAdd(a + 3, v.w);
    if (lane == 0) unsafeAtomicAdd(&deg[dst], 1.0f);
}

// ---------- 2) h = agg/max(deg,1) (in place); hn16 = f16(h / (||h||+eps)) ----------
__global__ void __launch_bounds__(256)
norm_kernel(float* __restrict__ h, const float* __restrict__ deg,
            half_t* __restrict__ hn16) {
    int wave = (blockIdx.x * blockDim.x + threadIdx.x) >> 5;
    int lane = threadIdx.x & 31;
    if (wave >= N_NODES) return;
    float d = deg[wave];
    d = d > 1.0f ? d : 1.0f;
    float inv = 1.0f / d;
    float4 v = *(float4*)(h + (size_t)wave * D + lane * 4);
    v.x *= inv; v.y *= inv; v.z *= inv; v.w *= inv;
    *(float4*)(h + (size_t)wave * D + lane * 4) = v;
    float ss = v.x * v.x + v.y * v.y + v.z * v.z + v.w * v.w;
    #pragma unroll
    for (int off = 16; off; off >>= 1) ss += __shfl_xor(ss, off, 32);
    float rn = 1.0f / (sqrtf(ss) + EPS);
    half_t* o = hn16 + (size_t)wave * D + lane * 4;
    o[0] = (half_t)(v.x * rn);
    o[1] = (half_t)(v.y * rn);
    o[2] = (half_t)(v.z * rn);
    o[3] = (half_t)(v.w * rn);
}

// ---------- 3) fused sim-GEMM (hn @ hn[:1024].T) + row argmax ----------
// Block = 256 thr (8 waves) handles a 32-row strip (two 16-row M-tiles) x all
// 1024 cols. Wave w handles N-tiles w*8 .. w*8+7; each B-tile set is reused by
// both M-tiles (halves centroid L2 traffic vs 16-row blocks). K=128 = 4 WMMAs
// per (M-tile, N-tile), chained through C.
__global__ void __launch_bounds__(256)
sim_argmax_kernel(const half_t* __restrict__ hn16, int* __restrict__ assign) {
    __shared__ float sval[32][8];
    __shared__ int   sidx[32][8];
    int lane   = threadIdx.x & 31;
    int waveId = threadIdx.x >> 5;
    int m0 = blockIdx.x * 32;       // tile 0: rows m0..m0+15 (always in range)
    int m1 = m0 + 16;               // tile 1: may run past N_NODES in last block

    int rowA0 = m0 + (lane & 15);
    int rowA1 = m1 + (lane & 15);
    if (rowA1 > N_NODES - 1) rowA1 = N_NODES - 1;   // clamp loads; stores guarded

    v16h a0[4], a1[4];
    #pragma unroll
    for (int kt = 0; kt < 4; kt++) {
        a0[kt] = load_tileA_row(hn16, D, rowA0, kt * 32, lane);
        a1[kt] = load_tileA_row(hn16, D, rowA1, kt * 32, lane);
    }

    float bv0[8], bv1[8];
    int   bi0[8], bi1[8];
    #pragma unroll
    for (int j = 0; j < 8; j++) {
        bv0[j] = -3.0e38f; bi0[j] = 0;
        bv1[j] = -3.0e38f; bi1[j] = 0;
    }

    for (int t = 0; t < 8; t++) {
        int n0 = (waveId * 8 + t) * 16;
        v16h b0 = load_tileB32x16(hn16, D, n0,  0, lane);
        v16h b1 = load_tileB32x16(hn16, D, n0, 32, lane);
        v16h b2 = load_tileB32x16(hn16, D, n0, 64, lane);
        v16h b3 = load_tileB32x16(hn16, D, n0, 96, lane);
        v8f c0 = {};
        v8f c1 = {};
        c0 = __builtin_amdgcn_wmma_f32_16x16x32_f16(false, a0[0], false, b0, (short)0, c0, false, false);
        c1 = __builtin_amdgcn_wmma_f32_16x16x32_f16(false, a1[0], false, b0, (short)0, c1, false, false);
        c0 = __builtin_amdgcn_wmma_f32_16x16x32_f16(false, a0[1], false, b1, (short)0, c0, false, false);
        c1 = __builtin_amdgcn_wmma_f32_16x16x32_f16(false, a1[1], false, b1, (short)0, c1, false, false);
        c0 = __builtin_amdgcn_wmma_f32_16x16x32_f16(false, a0[2], false, b2, (short)0, c0, false, false);
        c1 = __builtin_amdgcn_wmma_f32_16x16x32_f16(false, a1[2], false, b2, (short)0, c1, false, false);
        c0 = __builtin_amdgcn_wmma_f32_16x16x32_f16(false, a0[3], false, b3, (short)0, c0, false, false);
        c1 = __builtin_amdgcn_wmma_f32_16x16x32_f16(false, a1[3], false, b3, (short)0, c1, false, false);
        // D layout: c[j] = sim[mtile + j + 8*(lane>>4)][n0 + (lane&15)]
        int nidx = n0 + (lane & 15);
        #pragma unroll
        for (int j = 0; j < 8; j++) {
            float v = c0[j];
            if (v > bv0[j] || (v == bv0[j] && nidx < bi0[j])) { bv0[j] = v; bi0[j] = nidx; }
            float w = c1[j];
            if (w > bv1[j] || (w == bv1[j] && nidx < bi1[j])) { bv1[j] = w; bi1[j] = nidx; }
        }
    }
    // reduce (val,idx) across the 16 lanes of each half-wave (rows differ per half)
    #pragma unroll
    for (int off = 8; off; off >>= 1) {
        #pragma unroll
        for (int j = 0; j < 8; j++) {
            float v2 = __shfl_xor(bv0[j], off, 32);
            int   i2 = __shfl_xor(bi0[j], off, 32);
            if (v2 > bv0[j] || (v2 == bv0[j] && i2 < bi0[j])) { bv0[j] = v2; bi0[j] = i2; }
            float w2 = __shfl_xor(bv1[j], off, 32);
            int   k2 = __shfl_xor(bi1[j], off, 32);
            if (w2 > bv1[j] || (w2 == bv1[j] && k2 < bi1[j])) { bv1[j] = w2; bi1[j] = k2; }
        }
    }
    if ((lane & 15) == 0) {
        int rbase = (lane >> 4) * 8;
        #pragma unroll
        for (int j = 0; j < 8; j++) {
            sval[rbase + j][waveId]      = bv0[j];
            sidx[rbase + j][waveId]      = bi0[j];
            sval[16 + rbase + j][waveId] = bv1[j];
            sidx[16 + rbase + j][waveId] = bi1[j];
        }
    }
    __syncthreads();
    if (threadIdx.x < 32) {
        float bv = sval[threadIdx.x][0];
        int   bi = sidx[threadIdx.x][0];
        #pragma unroll
        for (int w = 1; w < 8; w++) {
            float v = sval[threadIdx.x][w];
            int   i = sidx[threadIdx.x][w];
            if (v > bv || (v == bv && i < bi)) { bv = v; bi = i; }
        }
        int row = m0 + threadIdx.x;
        if (row < N_NODES) assign[row] = bi;
    }
}

// ---------- 4) per-cluster sums: counts[a]+=1, xc[a]+=h[i] ----------
__global__ void __launch_bounds__(256)
cluster_agg_kernel(const float* __restrict__ h, const int* __restrict__ assign,
                   float* __restrict__ counts, float* __restrict__ xc) {
    int wave = (blockIdx.x * blockDim.x + threadIdx.x) >> 5;
    int lane = threadIdx.x & 31;
    if (wave >= N_NODES) return;
    int a = assign[wave];
    if (lane == 0) unsafeAtomicAdd(&counts[a], 1.0f);
    float4 v = *(const float4*)(h + (size_t)wave * D + lane * 4);
    float* p = xc + (size_t)a * D + lane * 4;
    unsafeAtomicAdd(p + 0, v.x);
    unsafeAtomicAdd(p + 1, v.y);
    unsafeAtomicAdd(p + 2, v.z);
    unsafeAtomicAdd(p + 3, v.w);
}

// ---------- 5) xcn16 = f16(xc/max(counts,1)); w116 = f16(W1) ----------
__global__ void __launch_bounds__(256)
prep_kernel(const float* __restrict__ xc, const float* __restrict__ counts,
            const float* __restrict__ W1,
            half_t* __restrict__ xcn16, half_t* __restrict__ w116) {
    int tid = blockIdx.x * blockDim.x + threadIdx.x;
    if (tid < K_CLUSTERS * D) {
        int m = tid / D;
        float c = counts[m];
        c = c > 1.0f ? c : 1.0f;
        xcn16[tid] = (half_t)(xc[tid] / c);
    }
    if (tid < D * D) w116[tid] = (half_t)W1[tid];
}

// ---------- 6) y = xcn @ W1.T + b1 (1024x128x128) via WMMA ----------
__global__ void __launch_bounds__(256)
out_gemm_kernel(const half_t* __restrict__ xcn16, const half_t* __restrict__ w116,
                const float* __restrict__ b1, float* __restrict__ y) {
    int lane   = threadIdx.x & 31;
    int waveId = threadIdx.x >> 5;
    int m0 = blockIdx.x * 16;
    int n0 = waveId * 16;
    int rowA = m0 + (lane & 15);
    v8f c = {};
    #pragma unroll
    for (int kt = 0; kt < 4; kt++) {
        v16h a = load_tileA_row(xcn16, D, rowA, kt * 32, lane);
        v16h b = load_tileB32x16(w116, D, n0, kt * 32, lane);
        c = __builtin_amdgcn_wmma_f32_16x16x32_f16(false, a, false, b, (short)0, c, false, false);
    }
    int n = n0 + (lane & 15);
    float bias = b1[n];
    int mbase = m0 + ((lane >> 4) << 3);
    #pragma unroll
    for (int j = 0; j < 8; j++)
        y[(size_t)(mbase + j) * D + n] = c[j] + bias;
}

// ---------- 7) out[i] = y[assign[i]] ----------
__global__ void __launch_bounds__(256)
gather_kernel(const float* __restrict__ y, const int* __restrict__ assign,
              float* __restrict__ out) {
    int tid = blockIdx.x * blockDim.x + threadIdx.x;
    if (tid >= N_NODES * 32) return;
    int node = tid >> 5;
    int l4   = (tid & 31) * 4;
    int a = assign[node];
    *(float4*)(out + (size_t)node * D + l4) =
        *(const float4*)(y + (size_t)a * D + l4);
}

extern "C" void kernel_launch(void* const* d_in, const int* in_sizes, int n_in,
                              void* d_out, int out_size, void* d_ws, size_t ws_size,
                              hipStream_t stream) {
    const float* x  = (const float*)d_in[0];
    const int*   ei = (const int*)d_in[1];
    const float* W1 = (const float*)d_in[2];
    const float* b1 = (const float*)d_in[3];
    float* out = (float*)d_out;

    char* ws = (char*)d_ws;
    size_t off = 0;
    auto take = [&](size_t bytes) {
        char* p = ws + off;
        off = (off + bytes + 255) & ~(size_t)255;
        return p;
    };
    float*  agg    = (float*)take((size_t)N_NODES * D * 4);   // becomes h in place
    float*  deg    = (float*)take((size_t)N_NODES * 4);
    half_t* hn16   = (half_t*)take((size_t)N_NODES * D * 2);
    int*    assign = (int*)take((size_t)N_NODES * 4);
    float*  counts = (float*)take((size_t)K_CLUSTERS * 4);
    float*  xc     = (float*)take((size_t)K_CLUSTERS * D * 4);
    float*  y      = (float*)take((size_t)K_CLUSTERS * D * 4);
    half_t* xcn16  = (half_t*)take((size_t)K_CLUSTERS * D * 2);
    half_t* w116   = (half_t*)take((size_t)D * D * 2);
    (void)ws_size; (void)in_sizes; (void)n_in; (void)out_size;

    hipMemsetAsync(agg,    0, (size_t)N_NODES * D * 4, stream);
    hipMemsetAsync(deg,    0, (size_t)N_NODES * 4, stream);
    hipMemsetAsync(counts, 0, (size_t)K_CLUSTERS * 4, stream);
    hipMemsetAsync(xc,     0, (size_t)K_CLUSTERS * D * 4, stream);

    agg_kernel<<<(N_EDGES * 32 + 255) / 256, 256, 0, stream>>>(x, ei, agg, deg);
    norm_kernel<<<(N_NODES * 32 + 255) / 256, 256, 0, stream>>>(agg, deg, hn16);
    sim_argmax_kernel<<<(N_NODES + 31) / 32, 256, 0, stream>>>(hn16, assign);
    cluster_agg_kernel<<<(N_NODES * 32 + 255) / 256, 256, 0, stream>>>(agg, assign, counts, xc);
    prep_kernel<<<(K_CLUSTERS * D + 255) / 256, 256, 0, stream>>>(xc, counts, W1, xcn16, w116);
    out_gemm_kernel<<<K_CLUSTERS / 16, 256, 0, stream>>>(xcn16, w116, b1, y);
    gather_kernel<<<(N_NODES * 32 + 255) / 256, 256, 0, stream>>>(y, assign, out);
}
